// BNNHAN_11038065951338
// MI455X (gfx1250) — compile-verified, matrix operand order checked by reading, not run on hardware
//
#include <hip/hip_runtime.h>
#include <math.h>

// ---------- WMMA types (gfx1250, wave32) ----------
typedef __attribute__((ext_vector_type(16))) __bf16 v16bf;
typedef __attribute__((ext_vector_type(8)))  float  v8f;

#define NEG_SLOPE 0.2f
#define H 8
#define DH 16
#define F_OUT 128
#define DIN 64

// ---------------- fragment loaders ----------------
// A-matrix 16x32 bf16 fragment from row-major f32 matrix (lda = row stride, multiple of 8).
// lane<16: row=lane&15, kb=k0;  lane>=16: same row, kb=k0+8.
// element e<8 -> K=kb+e ; e>=8 -> K=kb+8+e  (i.e. kb+16..kb+23).
__device__ inline v16bf load_a_frag(const float* __restrict__ A, int lda,
                                    int row, int k0, int lane, bool relu) {
  int kb = k0 + ((lane >> 4) & 1) * 8;
  const float4* p = reinterpret_cast<const float4*>(A + (size_t)row * lda + kb);
  float4 q0 = p[0];   // K = kb+0..3
  float4 q1 = p[1];   // K = kb+4..7
  float4 q2 = p[4];   // K = kb+16..19
  float4 q3 = p[5];   // K = kb+20..23
  float v[16] = {q0.x, q0.y, q0.z, q0.w, q1.x, q1.y, q1.z, q1.w,
                 q2.x, q2.y, q2.z, q2.w, q3.x, q3.y, q3.z, q3.w};
  v16bf a;
#pragma unroll
  for (int e = 0; e < 16; ++e) {
    float t = v[e];
    if (relu) t = fmaxf(t, 0.f);
    a[e] = (__bf16)t;
  }
  return a;
}

// B-matrix 32x16 bf16 fragment from row-major f32 weight [K x N] (ldb = N).
__device__ inline v16bf load_b_frag(const float* __restrict__ B, int ldb,
                                    int k0, int col, int lane) {
  int kb = k0 + ((lane >> 4) & 1) * 8;
  v16bf b;
#pragma unroll
  for (int e = 0; e < 8; ++e) {
    b[e]     = (__bf16)B[(size_t)(kb + e) * ldb + col];
    b[e + 8] = (__bf16)B[(size_t)(kb + 16 + e) * ldb + col];
  }
  return b;
}

// ---------------- init ----------------
__global__ void han_fill_f32(float* p, float v, size_t n) {
  size_t i = (size_t)blockIdx.x * blockDim.x + threadIdx.x;
  size_t stride = (size_t)gridDim.x * blockDim.x;
  for (; i < n; i += stride) p[i] = v;
}
__global__ void han_fill_u32(unsigned* p, unsigned v, size_t n) {
  size_t i = (size_t)blockIdx.x * blockDim.x + threadIdx.x;
  size_t stride = (size_t)gridDim.x * blockDim.x;
  for (; i < n; i += stride) p[i] = v;
}

// ---------------- projection GEMM: Hout = X[N x 64] @ W[64 x 128] + b ----------------
// one wave per 16-row strip; A fragments loaded once, 8 column tiles x 2 K-steps = 16 WMMA.
__global__ void __launch_bounds__(32) han_proj_wmma(
    const float* __restrict__ X, const float* __restrict__ W,
    const float* __restrict__ bias, float* __restrict__ Hout, int N) {
  int lane = threadIdx.x & 31;
  int row0 = blockIdx.x * 16;
  int rowA = row0 + (lane & 15);
  if (rowA >= N) rowA = N - 1;                 // clamp: EXEC stays all-1s for WMMA
  bool full = (row0 + 16 <= N);                // wave-uniform
  int rbase = row0 + ((lane >> 4) ? 8 : 0);

  v16bf a0 = load_a_frag(X, DIN, rowA, 0, lane, false);
  v16bf a1 = load_a_frag(X, DIN, rowA, 32, lane, false);

#pragma unroll
  for (int t = 0; t < F_OUT / 16; ++t) {
    int col = t * 16 + (lane & 15);
    v16bf b0 = load_b_frag(W, F_OUT, 0, col, lane);
    v16bf b1 = load_b_frag(W, F_OUT, 32, col, lane);
    v8f c = {};
    c = __builtin_amdgcn_wmma_f32_16x16x32_bf16(false, a0, false, b0, (short)0, c, false, false);
    c = __builtin_amdgcn_wmma_f32_16x16x32_bf16(false, a1, false, b1, (short)0, c, false, false);
    float bv = bias[col];
    if (full) {
#pragma unroll
      for (int r = 0; r < 8; ++r)
        Hout[(size_t)(rbase + r) * F_OUT + col] = c[r] + bv;
    } else {
#pragma unroll
      for (int r = 0; r < 8; ++r) {
        int rr = rbase + r;
        if (rr < N) Hout[(size_t)rr * F_OUT + col] = c[r] + bv;
      }
    }
  }
}

// ---------------- per-node per-head attention logits ----------------
__global__ void han_att_logits(const float* __restrict__ Hm, const float* __restrict__ att,
                               float* __restrict__ out, int N) {
  int i = blockIdx.x * blockDim.x + threadIdx.x;
  if (i >= N * H) return;
  int h = i & (H - 1);
  int n = i >> 3;
  const float4* hp = reinterpret_cast<const float4*>(Hm + (size_t)n * F_OUT + h * DH);
  const float4* ap = reinterpret_cast<const float4*>(att + h * DH);
  float s = 0.f;
#pragma unroll
  for (int d = 0; d < 4; ++d) {
    float4 hv = hp[d], av = ap[d];
    s += hv.x * av.x + hv.y * av.y + hv.z * av.z + hv.w * av.w;
  }
  out[i] = s;
}

// -------- order-preserving float<->uint encode for atomicMax over signed floats --------
__device__ inline unsigned encf(float f) {
  unsigned u = __float_as_uint(f);
  return (u & 0x80000000u) ? ~u : (u | 0x80000000u);
}
__device__ inline float decf(unsigned u) {
  unsigned b = (u & 0x80000000u) ? (u & 0x7FFFFFFFu) : ~u;
  return __uint_as_float(b);
}
#define ENC_NEG_INF 0x007FFFFFu   // encf(-inf)

__device__ inline float lrelu(float a) { return (a > 0.f) ? a : NEG_SLOPE * a; }

// ---------------- edge softmax pass 1: segment max ----------------
__global__ void han_edge_max(const int* __restrict__ src, const int* __restrict__ dst,
                             const float* __restrict__ asrc, const float* __restrict__ adst,
                             unsigned* __restrict__ mx, int E) {
  int e = blockIdx.x * blockDim.x + threadIdx.x;
  if (e >= E) return;
  int s = src[e], d = dst[e];
#pragma unroll
  for (int h = 0; h < H; ++h) {
    float a = lrelu(asrc[(size_t)s * H + h] + adst[(size_t)d * H + h]);
    atomicMax(&mx[(size_t)d * H + h], encf(a));
  }
}

// ---------------- edge softmax pass 2: segment sum of exp ----------------
__global__ void han_edge_sum(const int* __restrict__ src, const int* __restrict__ dst,
                             const float* __restrict__ asrc, const float* __restrict__ adst,
                             const unsigned* __restrict__ mx, float* __restrict__ sm, int E) {
  int e = blockIdx.x * blockDim.x + threadIdx.x;
  if (e >= E) return;
  int s = src[e], d = dst[e];
#pragma unroll
  for (int h = 0; h < H; ++h) {
    float a = lrelu(asrc[(size_t)s * H + h] + adst[(size_t)d * H + h]);
    float ex = __expf(a - decf(mx[(size_t)d * H + h]));
    atomicAdd(&sm[(size_t)d * H + h], ex);
  }
}

// ---------------- edge pass 3: weighted message scatter ----------------
// one thread per (edge, head): vector gather + 16 float atomics into o[dst]
__global__ void han_edge_msg(const int* __restrict__ src, const int* __restrict__ dst,
                             const float* __restrict__ asrc, const float* __restrict__ adst,
                             const unsigned* __restrict__ mx, const float* __restrict__ sm,
                             const float* __restrict__ Hsrc, float* __restrict__ O, int E) {
  long long t = (long long)blockIdx.x * blockDim.x + threadIdx.x;
  if (t >= (long long)E * H) return;
  int h = (int)(t & (H - 1));
  int e = (int)(t >> 3);
  int s = src[e], d = dst[e];
  float a = lrelu(asrc[(size_t)s * H + h] + adst[(size_t)d * H + h]);
  float w = __expf(a - decf(mx[(size_t)d * H + h])) / (sm[(size_t)d * H + h] + 1e-16f);
  const float4* hp = reinterpret_cast<const float4*>(Hsrc + (size_t)s * F_OUT + h * DH);
  float* op = O + (size_t)d * F_OUT + h * DH;
#pragma unroll
  for (int v = 0; v < 4; ++v) {
    float4 hv = hp[v];
    atomicAdd(&op[v * 4 + 0], w * hv.x);
    atomicAdd(&op[v * 4 + 1], w * hv.y);
    atomicAdd(&op[v * 4 + 2], w * hv.z);
    atomicAdd(&op[v * 4 + 3], w * hv.w);
  }
}

// ---------------- semantic attention score (fused GEMM epilogue) ----------------
// one wave per 16-row strip of relu(O) @ Kw : 4 K-steps x 8 N-tiles = 32 WMMA,
// 8 persistent f32 accumulators; epilogue tanh * q reduced straight to score[m].
__global__ void __launch_bounds__(32) han_sem_score_wmma(
    const float* __restrict__ O, const float* __restrict__ Kw,
    const float* __restrict__ kb, const float* __restrict__ q,
    float* __restrict__ score, int m, int N) {
  int lane = threadIdx.x & 31;
  int row0 = blockIdx.x * 16;
  int rowA = row0 + (lane & 15);
  if (rowA >= N) rowA = N - 1;                 // clamp for EXEC; masked in epilogue
  int rbase = row0 + ((lane >> 4) ? 8 : 0);
  bool full = (row0 + 16 <= N);                // wave-uniform

  v8f acc[F_OUT / 16];
#pragma unroll
  for (int t = 0; t < F_OUT / 16; ++t) acc[t] = (v8f){};

#pragma unroll
  for (int k0 = 0; k0 < F_OUT; k0 += 32) {
    v16bf a = load_a_frag(O, F_OUT, rowA, k0, lane, /*relu=*/true);
#pragma unroll
    for (int t = 0; t < F_OUT / 16; ++t) {
      v16bf b = load_b_frag(Kw, F_OUT, k0, t * 16 + (lane & 15), lane);
      acc[t] = __builtin_amdgcn_wmma_f32_16x16x32_bf16(false, a, false, b, (short)0,
                                                       acc[t], false, false);
    }
  }

  float p = 0.f;
#pragma unroll
  for (int t = 0; t < F_OUT / 16; ++t) {
    int col = t * 16 + (lane & 15);
    float kbv = kb[col], qv = q[col];
    if (full) {
#pragma unroll
      for (int r = 0; r < 8; ++r) p += tanhf(acc[t][r] + kbv) * qv;
    } else {
#pragma unroll
      for (int r = 0; r < 8; ++r)
        if (rbase + r < N) p += tanhf(acc[t][r] + kbv) * qv;
    }
  }
  // wave32 reduce
#pragma unroll
  for (int off = 16; off > 0; off >>= 1) p += __shfl_down(p, off, 32);
  if (lane == 0) atomicAdd(&score[m], p);
}

// ---------------- 2-way softmax over metapath scores ----------------
__global__ void han_sem_softmax(const float* __restrict__ score, float* __restrict__ attn,
                                float invN) {
  float s0 = score[0] * invN, s1 = score[1] * invN;
  float mx = fmaxf(s0, s1);
  float e0 = __expf(s0 - mx), e1 = __expf(s1 - mx);
  float inv = 1.f / (e0 + e1);
  attn[0] = e0 * inv;
  attn[1] = e1 * inv;
}

// ---------------- fused combine + output linear [128 x 2] ----------------
__global__ void han_final(const float* __restrict__ o1, const float* __restrict__ o2,
                          const float* __restrict__ attn, const float* __restrict__ lw,
                          const float* __restrict__ lb, float* __restrict__ out, int N) {
  int n = blockIdx.x * blockDim.x + threadIdx.x;
  if (n >= N) return;
  float a0 = attn[0], a1 = attn[1];
  float acc0 = lb[0], acc1 = lb[1];
  const float4* p1 = reinterpret_cast<const float4*>(o1 + (size_t)n * F_OUT);
  const float4* p2 = reinterpret_cast<const float4*>(o2 + (size_t)n * F_OUT);
#pragma unroll 8
  for (int v = 0; v < F_OUT / 4; ++v) {
    float4 u1 = p1[v], u2 = p2[v];
    float f0 = a0 * fmaxf(u1.x, 0.f) + a1 * fmaxf(u2.x, 0.f);
    float f1 = a0 * fmaxf(u1.y, 0.f) + a1 * fmaxf(u2.y, 0.f);
    float f2 = a0 * fmaxf(u1.z, 0.f) + a1 * fmaxf(u2.z, 0.f);
    float f3 = a0 * fmaxf(u1.w, 0.f) + a1 * fmaxf(u2.w, 0.f);
    int f = v * 4;
    acc0 += f0 * lw[f * 2 + 0] + f1 * lw[f * 2 + 2] + f2 * lw[f * 2 + 4] + f3 * lw[f * 2 + 6];
    acc1 += f0 * lw[f * 2 + 1] + f1 * lw[f * 2 + 3] + f2 * lw[f * 2 + 5] + f3 * lw[f * 2 + 7];
  }
  out[n * 2 + 0] = acc0;
  out[n * 2 + 1] = acc1;
}

// ---------------- launch ----------------
extern "C" void kernel_launch(void* const* d_in, const int* in_sizes, int n_in,
                              void* d_out, int out_size, void* d_ws, size_t ws_size,
                              hipStream_t stream) {
  const float* x_subj   = (const float*)d_in[0];
  const float* x_chan   = (const float*)d_in[1];
  const int*   edge_cs  = (const int*)d_in[2];
  const int*   edge_ss  = (const int*)d_in[3];
  const float* W_subj   = (const float*)d_in[4];
  const float* b_subj   = (const float*)d_in[5];
  const float* W_chan   = (const float*)d_in[6];
  const float* b_chan   = (const float*)d_in[7];
  const float* att_s_cs = (const float*)d_in[8];
  const float* att_d_cs = (const float*)d_in[9];
  const float* att_s_ss = (const float*)d_in[10];
  const float* att_d_ss = (const float*)d_in[11];
  const float* k_w      = (const float*)d_in[12];
  const float* k_b      = (const float*)d_in[13];
  const float* q        = (const float*)d_in[14];
  const float* lin_w    = (const float*)d_in[15];
  const float* lin_b    = (const float*)d_in[16];
  float* out = (float*)d_out;

  const int Ns  = in_sizes[0] / DIN;
  const int Nc  = in_sizes[1] / DIN;
  const int Ecs = in_sizes[2] / 2;
  const int Ess = in_sizes[3] / 2;

  // ---- workspace carve-up (floats) ----
  float* ws = (float*)d_ws;
  size_t off = 0;
  float*    hs    = ws + off; off += (size_t)Ns * F_OUT;
  float*    hc    = ws + off; off += (size_t)Nc * F_OUT;
  float*    as_cs = ws + off; off += (size_t)Nc * H;
  float*    ad_cs = ws + off; off += (size_t)Ns * H;
  float*    as_ss = ws + off; off += (size_t)Ns * H;
  float*    ad_ss = ws + off; off += (size_t)Ns * H;
  unsigned* mx_cs = (unsigned*)(ws + off); off += (size_t)Ns * H;
  unsigned* mx_ss = (unsigned*)(ws + off); off += (size_t)Ns * H;
  float*    sm_cs = ws + off; off += (size_t)Ns * H;
  float*    sm_ss = ws + off; off += (size_t)Ns * H;
  float*    o1    = ws + off; off += (size_t)Ns * F_OUT;
  float*    o2    = ws + off; off += (size_t)Ns * F_OUT;
  float*    score = ws + off; off += 2;
  float*    attn  = ws + off; off += 2;

  const int* src_cs = edge_cs;          const int* dst_cs = edge_cs + Ecs;
  const int* src_ss = edge_ss;          const int* dst_ss = edge_ss + Ess;

  // ---- init accumulators (re-done every call; harness does not re-zero) ----
  {
    han_fill_u32<<<1024, 256, 0, stream>>>(mx_cs, ENC_NEG_INF, (size_t)Ns * H * 2);
    han_fill_f32<<<1024, 256, 0, stream>>>(sm_cs, 0.f, (size_t)Ns * H * 2);
    han_fill_f32<<<2048, 256, 0, stream>>>(o1, 0.f, (size_t)Ns * F_OUT * 2 + 2);
  }

  // ---- 1) projections via WMMA (one wave per 16-row strip) ----
  han_proj_wmma<<<(Ns + 15) / 16, 32, 0, stream>>>(x_subj, W_subj, b_subj, hs, Ns);
  han_proj_wmma<<<(Nc + 15) / 16, 32, 0, stream>>>(x_chan, W_chan, b_chan, hc, Nc);

  // ---- 2) per-node attention logits ----
  {
    int tc = 256;
    han_att_logits<<<(Nc * H + tc - 1) / tc, tc, 0, stream>>>(hc, att_s_cs, as_cs, Nc);
    han_att_logits<<<(Ns * H + tc - 1) / tc, tc, 0, stream>>>(hs, att_d_cs, ad_cs, Ns);
    han_att_logits<<<(Ns * H + tc - 1) / tc, tc, 0, stream>>>(hs, att_s_ss, as_ss, Ns);
    han_att_logits<<<(Ns * H + tc - 1) / tc, tc, 0, stream>>>(hs, att_d_ss, ad_ss, Ns);
  }

  // ---- 3) edge softmax (3 scatter passes per edge type) ----
  {
    int tc = 256;
    han_edge_max<<<(Ecs + tc - 1) / tc, tc, 0, stream>>>(src_cs, dst_cs, as_cs, ad_cs, mx_cs, Ecs);
    han_edge_max<<<(Ess + tc - 1) / tc, tc, 0, stream>>>(src_ss, dst_ss, as_ss, ad_ss, mx_ss, Ess);
    han_edge_sum<<<(Ecs + tc - 1) / tc, tc, 0, stream>>>(src_cs, dst_cs, as_cs, ad_cs, mx_cs, sm_cs, Ecs);
    han_edge_sum<<<(Ess + tc - 1) / tc, tc, 0, stream>>>(src_ss, dst_ss, as_ss, ad_ss, mx_ss, sm_ss, Ess);
    long long ncs = (long long)Ecs * H, nss = (long long)Ess * H;
    han_edge_msg<<<(unsigned)((ncs + tc - 1) / tc), tc, 0, stream>>>(
        src_cs, dst_cs, as_cs, ad_cs, mx_cs, sm_cs, hc, o1, Ecs);
    han_edge_msg<<<(unsigned)((nss + tc - 1) / tc), tc, 0, stream>>>(
        src_ss, dst_ss, as_ss, ad_ss, mx_ss, sm_ss, hs, o2, Ess);
  }

  // ---- 4) semantic attention: fused WMMA GEMM -> tanh -> q-dot -> global reduce ----
  {
    han_sem_score_wmma<<<(Ns + 15) / 16, 32, 0, stream>>>(o1, k_w, k_b, q, score, 0, Ns);
    han_sem_score_wmma<<<(Ns + 15) / 16, 32, 0, stream>>>(o2, k_w, k_b, q, score, 1, Ns);
    han_sem_softmax<<<1, 1, 0, stream>>>(score, attn, 1.0f / (float)Ns);
  }

  // ---- 5) fused combine + output linear ----
  han_final<<<(Ns + 255) / 256, 256, 0, stream>>>(o1, o2, attn, lin_w, lin_b, out, Ns);
}
